// SwinTransformer_26792005992988
// MI455X (gfx1250) — matrix operand order repaired and checked by reading, not compiled, not run on hardware
//
#include <hip/hip_runtime.h>
#include <hip/hip_bf16.h>
#include <math.h>

// ---------------------------------------------------------------------------
// Swin-Tiny forward for MI455X (gfx1250, wave32).
// GEMMs: bf16 WMMA (v_wmma_f32_16x16x32_bf16, fp32 accum).
//   - Weights repacked per-launch into the exact ISA B-fragment layout
//     (32 contiguous bytes/lane -> 2x global_load_b128, no guards).
//   - Activations (A operand) produced in bf16 row-major: each lane reads
//     two contiguous 8x bf16 runs -> 2x global_load_b128.
//   - Dual accumulators: two independent WMMA dependency chains per wave.
// Attention (49x49, hd=32): VALU + LDS. Shifts/windows/merge: index math.
// ---------------------------------------------------------------------------

typedef __attribute__((ext_vector_type(16))) __bf16 v16bf;
typedef __attribute__((ext_vector_type(8)))  __bf16 v8bf;
typedef __attribute__((ext_vector_type(8)))  float  v8f;

#define WAVE 32

static inline int cdiv(long long a, long long b) { return (int)((a + b - 1) / b); }

// ---------------------------------------------------------------------------
// Patch embed: conv 4x4 stride 4, NCHW in, token-major fp32 [B,56*56,96] out.
// ---------------------------------------------------------------------------
__global__ void patch_embed_kernel(const float* __restrict__ x,
                                   const float* __restrict__ w,
                                   const float* __restrict__ bias,
                                   float* __restrict__ out) {
  long long idx = (long long)blockIdx.x * blockDim.x + threadIdx.x;
  const long long total = 32LL * 56 * 56 * 96;
  if (idx >= total) return;
  int c   = (int)(idx % 96);
  long long tok = idx / 96;
  int px  = (int)(tok % 56);
  int py  = (int)((tok / 56) % 56);
  int b   = (int)(tok / (56 * 56));
  float acc = bias[c];
  #pragma unroll
  for (int i = 0; i < 3; ++i)
    for (int k = 0; k < 4; ++k)
      for (int l = 0; l < 4; ++l) {
        float xv = x[(((long long)b * 3 + i) * 224 + (py * 4 + k)) * 224 + (px * 4 + l)];
        float wv = w[((c * 3 + i) * 4 + k) * 4 + l];
        acc = fmaf(xv, wv, acc);
      }
  out[((long long)b * 3136 + (long long)py * 56 + px) * 96 + c] = acc;
}

// ---------------------------------------------------------------------------
// LayerNorm: one wave per row. out_bf16 selects fp32 or bf16 output.
// ---------------------------------------------------------------------------
__global__ void ln_kernel(const float* __restrict__ in, void* __restrict__ out,
                          const float* __restrict__ g, const float* __restrict__ b,
                          int C, int out_bf16) {
  long long row = blockIdx.x;
  const float* xr = in + row * (long long)C;
  float s = 0.f, s2 = 0.f;
  for (int i = threadIdx.x; i < C; i += WAVE) {
    float v = xr[i];
    s += v; s2 += v * v;
  }
  #pragma unroll
  for (int off = 16; off > 0; off >>= 1) {
    s  += __shfl_xor(s,  off, WAVE);
    s2 += __shfl_xor(s2, off, WAVE);
  }
  float mean = s / C;
  float var  = s2 / C - mean * mean;
  float inv  = rsqrtf(var + 1e-5f);
  if (out_bf16) {
    __bf16* orow = (__bf16*)out + row * (long long)C;
    for (int i = threadIdx.x; i < C; i += WAVE)
      orow[i] = (__bf16)((xr[i] - mean) * inv * g[i] + b[i]);
  } else {
    float* orow = (float*)out + row * (long long)C;
    for (int i = threadIdx.x; i < C; i += WAVE)
      orow[i] = (xr[i] - mean) * inv * g[i] + b[i];
  }
}

// ---------------------------------------------------------------------------
// Weight repack: fp32 [K,N] -> bf16 WMMA B-fragment layout.
// Fragment addr = ((tile*ksteps + ks)*32 + lane)*16 bf16 (32B/lane contiguous).
// Element 2v(+1) of a lane = W[ks*32 + 2v(+1) + 8*(lane>>4) + 8*(v>=4)][tile*16+(lane&15)].
// Columns >= N are zero-padded (head N=1000 -> 63 tiles).
// ---------------------------------------------------------------------------
__global__ void repack_w_kernel(const float* __restrict__ W, __bf16* __restrict__ out,
                                int N, int ksteps, long long total) {
  long long idx = (long long)blockIdx.x * blockDim.x + threadIdx.x;
  if (idx >= total) return;        // total = ntiles*ksteps*32
  int lane = (int)(idx & 31);
  long long rest = idx >> 5;
  int ks   = (int)(rest % ksteps);
  int tile = (int)(rest / ksteps);
  int half = lane >> 4;
  int ncol = tile * 16 + (lane & 15);
  v16bf vals;
  #pragma unroll
  for (int v = 0; v < 8; ++v) {
    int k = ks * 32 + 2 * v + 8 * half + ((v >= 4) ? 8 : 0);
    float w0 = (ncol < N) ? W[(long long)k * N + ncol]       : 0.f;
    float w1 = (ncol < N) ? W[(long long)(k + 1) * N + ncol] : 0.f;
    vals[2 * v]     = (__bf16)w0;
    vals[2 * v + 1] = (__bf16)w1;
  }
  *(v16bf*)(out + idx * 16) = vals;
}

// ---------------------------------------------------------------------------
// WMMA GEMM: out[M,N] = A[M,K](bf16) @ Wrep (+bias)(gelu?)(+resid fp32).
// One wave per 16x16 tile. Inner loop: 4x b128 loads + 1 wmma, no branches.
// Two accumulators -> two independent WMMA RAW chains (hazard hiding).
// flags: bit0 = exact GELU, bit1 = bf16 output.
// ---------------------------------------------------------------------------
__device__ __forceinline__ v16bf load_a_frag(const __bf16* arow, int ks) {
  v8bf a_lo = *(const v8bf*)(arow + ks * 32);        // K = kk + half*8 .. +8
  v8bf a_hi = *(const v8bf*)(arow + ks * 32 + 16);   // K = kk+16+half*8 .. +8
  return __builtin_shufflevector(a_lo, a_hi,
      0, 1, 2, 3, 4, 5, 6, 7, 8, 9, 10, 11, 12, 13, 14, 15);
}

__global__ void gemm_wmma_kernel(const __bf16* __restrict__ A,
                                 const __bf16* __restrict__ Wrep,
                                 const float* __restrict__ bias,
                                 const float* __restrict__ resid,
                                 void* __restrict__ Cout,
                                 int M, int N, int K, int ksteps, int flags) {
  const int tile = blockIdx.x;          // N tile
  const int row0 = blockIdx.y * 16;     // M tile
  const int lane = threadIdx.x;
  const int half = lane >> 4;
  const int mrow = row0 + (lane & 15);
  const int ncol = tile * 16 + (lane & 15);

  const __bf16* arow  = A + (long long)mrow * K + half * 8;
  const __bf16* bfrag = Wrep + ((long long)tile * ksteps) * (32 * 16) + (long long)lane * 16;

  v8f acc0 = {}, acc1 = {};
  int ks = 0;
  for (; ks + 1 < ksteps; ks += 2) {
    v16bf af0 = load_a_frag(arow, ks);
    v16bf bf0 = *(const v16bf*)(bfrag + (long long)ks * (32 * 16));
    v16bf af1 = load_a_frag(arow, ks + 1);
    v16bf bf1 = *(const v16bf*)(bfrag + (long long)(ks + 1) * (32 * 16));
    acc0 = __builtin_amdgcn_wmma_f32_16x16x32_bf16(
        false, af0, false, bf0, (short)0, acc0, false, false);
    acc1 = __builtin_amdgcn_wmma_f32_16x16x32_bf16(
        false, af1, false, bf1, (short)0, acc1, false, false);
  }
  if (ks < ksteps) {
    v16bf af = load_a_frag(arow, ks);
    v16bf bf = *(const v16bf*)(bfrag + (long long)ks * (32 * 16));
    acc0 = __builtin_amdgcn_wmma_f32_16x16x32_bf16(
        false, af, false, bf, (short)0, acc0, false, false);
  }

  // C/D layout: lanes 0-15 -> M=r, lanes 16-31 -> M=8+r; N = lane&15.
  const bool colok = (ncol < N);
  #pragma unroll
  for (int r = 0; r < 8; ++r) {
    int grow = row0 + r + 8 * half;
    if (colok) {
      float v = acc0[r] + acc1[r];
      if (bias)      v += bias[ncol];
      if (flags & 1) v = 0.5f * v * (1.f + erff(v * 0.70710678118f)); // exact GELU
      if (resid)     v += resid[(long long)grow * N + ncol];
      if (flags & 2) ((__bf16*)Cout)[(long long)grow * N + ncol] = (__bf16)v;
      else           ((float*)Cout)[(long long)grow * N + ncol] = v;
    }
  }
}

// ---------------------------------------------------------------------------
// Window partition with cyclic shift fused; fp32 in -> bf16 out (GEMM A).
// ---------------------------------------------------------------------------
__global__ void win_gather_kernel(const float* __restrict__ in, __bf16* __restrict__ out,
                                  int H, int C, int shift, long long total) {
  long long idx = (long long)blockIdx.x * blockDim.x + threadIdx.x;
  if (idx >= total) return;
  int ch = (int)(idx % C);
  long long tok = idx / C;
  int t  = (int)(tok % 49);
  long long wi = tok / 49;
  int nw  = H / 7;
  int wim = (int)(wi % (nw * nw));
  int b   = (int)(wi / (nw * nw));
  int wr = wim / nw, wc = wim % nw;
  int pr = t / 7, pc = t % 7;
  int rr = wr * 7 + pr + shift; if (rr >= H) rr -= H;
  int cc = wc * 7 + pc + shift; if (cc >= H) cc -= H;
  out[idx] = (__bf16)in[(((long long)b * H + rr) * H + cc) * C + ch];
}

// ---------------------------------------------------------------------------
// Window reverse + un-shift + residual add (all fp32): out = sc + rev(win).
// ---------------------------------------------------------------------------
__global__ void win_scatter_add_kernel(const float* __restrict__ sc,
                                       const float* __restrict__ win,
                                       float* __restrict__ out,
                                       int H, int C, int shift, long long total) {
  long long idx = (long long)blockIdx.x * blockDim.x + threadIdx.x;
  if (idx >= total) return;
  int ch = (int)(idx % C);
  long long tok = idx / C;
  int b   = (int)(tok / ((long long)H * H));
  int rem = (int)(tok % ((long long)H * H));
  int r = rem / H, c = rem % H;
  int rr = r - shift; if (rr < 0) rr += H;
  int cc = c - shift; if (cc < 0) cc += H;
  int nw = H / 7;
  int wr = rr / 7, pr = rr % 7, wc = cc / 7, pc = cc % 7;
  long long srow = (((long long)b * nw + wr) * nw + wc) * 49 + pr * 7 + pc;
  out[idx] = sc[idx] + win[srow * C + ch];
}

// ---------------------------------------------------------------------------
// Window attention (hd == 32 for all stages). One block per (window, head).
// fp32 qkv in, bf16 out (feeds proj GEMM A).
// ---------------------------------------------------------------------------
__device__ __forceinline__ int region_id(int g, int H) {
  // slices: [0, H-7), [H-7, H-3), [H-3, H)   (ws=7, shift=3)
  return (g < H - 7) ? 0 : ((g < H - 3) ? 1 : 2);
}

__global__ void attn_kernel(const float* __restrict__ qkv,
                            const float* __restrict__ rel,
                            __bf16* __restrict__ out,
                            int nh, int C, int nw, int shift, int H) {
  const int wid = blockIdx.x / nh;
  const int h   = blockIdx.x % nh;
  const int wim = wid % (nw * nw);
  const int wr = wim / nw, wc = wim % nw;
  const int threeC = 3 * C;
  const long long base = (long long)wid * 49;

  __shared__ float qb[49][32];
  __shared__ float kb[49][32];
  __shared__ float vb[49][32];
  for (int i = threadIdx.x; i < 49 * 32; i += blockDim.x) {
    int t = i >> 5, d = i & 31;
    const float* row = qkv + (base + t) * threeC;
    qb[t][d] = row[h * 32 + d];
    kb[t][d] = row[C + h * 32 + d];
    vb[t][d] = row[2 * C + h * 32 + d];
  }
  __syncthreads();

  const int t = threadIdx.x;
  if (t < 49) {
    const float scale = 0.17677669529f; // 32^-0.5
    float q[32];
    #pragma unroll
    for (int d = 0; d < 32; ++d) q[d] = qb[t][d] * scale;

    const int qr = t / 7, qc = t % 7;
    const int rq = region_id(wr * 7 + qr, H) * 3 + region_id(wc * 7 + qc, H);

    float s[49];
    float mx = -3.4e38f;
    for (int j = 0; j < 49; ++j) {
      float d = 0.f;
      #pragma unroll
      for (int k = 0; k < 32; ++k) d = fmaf(q[k], kb[j][k], d);
      int kr = j / 7, kc = j % 7;
      int ridx = (qr - kr + 6) * 13 + (qc - kc + 6);
      d += rel[ridx * nh + h];
      if (shift > 0) {
        int rk = region_id(wr * 7 + kr, H) * 3 + region_id(wc * 7 + kc, H);
        if (rk != rq) d -= 100.f;
      }
      s[j] = d;
      mx = fmaxf(mx, d);
    }
    float sum = 0.f;
    float o[32];
    #pragma unroll
    for (int d = 0; d < 32; ++d) o[d] = 0.f;
    for (int j = 0; j < 49; ++j) {
      float p = __expf(s[j] - mx);
      sum += p;
      #pragma unroll
      for (int d = 0; d < 32; ++d) o[d] = fmaf(p, vb[j][d], o[d]);
    }
    float inv = 1.f / sum;
    __bf16* orow = out + (base + t) * C + h * 32;
    #pragma unroll
    for (int d = 0; d < 32; ++d) orow[d] = (__bf16)(o[d] * inv);
  }
}

// ---------------------------------------------------------------------------
// Patch merging gather (fp32): [B,H,H,C] -> [B, H/2*H/2, 4C].
// ---------------------------------------------------------------------------
__global__ void merge_gather_kernel(const float* __restrict__ in, float* __restrict__ out,
                                    int H, int C, long long total) {
  long long idx = (long long)blockIdx.x * blockDim.x + threadIdx.x;
  if (idx >= total) return;
  int C4 = 4 * C;
  int col = (int)(idx % C4);
  long long row = idx / C4;
  int half = H / 2;
  int b   = (int)(row / ((long long)half * half));
  int rem = (int)(row % ((long long)half * half));
  int r2 = rem / half, c2 = rem % half;
  int q  = col / C, ch = col % C;
  int dr = (q == 1 || q == 3) ? 1 : 0; // x0:(0,0) x1:(1,0) x2:(0,1) x3:(1,1)
  int dc = (q >= 2) ? 1 : 0;
  out[idx] = in[(((long long)b * H + (2 * r2 + dr)) * H + (2 * c2 + dc)) * C + ch];
}

// ---------------------------------------------------------------------------
// Mean over tokens: fp32 [32, L, C] -> bf16 [32, C] (feeds head GEMM A).
// ---------------------------------------------------------------------------
__global__ void mean_pool_kernel(const float* __restrict__ in, __bf16* __restrict__ out,
                                 int C, int L) {
  int idx = blockIdx.x * blockDim.x + threadIdx.x;
  if (idx >= 32 * C) return;
  int b = idx / C, c = idx % C;
  float s = 0.f;
  for (int t = 0; t < L; ++t) s += in[((long long)b * L + t) * C + c];
  out[idx] = (__bf16)(s / (float)L);
}

// ---------------------------------------------------------------------------
// Host orchestration.
// ---------------------------------------------------------------------------
extern "C" void kernel_launch(void* const* d_in, const int* in_sizes, int n_in,
                              void* d_out, int out_size, void* d_ws, size_t ws_size,
                              hipStream_t stream) {
  (void)in_sizes; (void)n_in; (void)out_size; (void)ws_size;

  static const int depths[4] = {2, 2, 6, 2};
  static const int dims[4]   = {96, 192, 384, 768};
  static const int res[4]    = {56, 28, 14, 7};
  static const int heads[4]  = {3, 6, 12, 24};
  static const int shifts[4] = {3, 3, 3, 0};

  // --- unpack inputs in setup_inputs() dict order -------------------------
  int p = 0;
  const float* x_in    = (const float*)d_in[p++];
  const float* patch_w = (const float*)d_in[p++];
  const float* patch_b = (const float*)d_in[p++];
  const float* pn_g    = (const float*)d_in[p++];
  const float* pn_b    = (const float*)d_in[p++];
  const float* bp[12][13]; // n1g n1b qkv_w qkv_b rel proj_w proj_b n2g n2b fc1_w fc1_b fc2_w fc2_b
  const float *mg[3], *mb[3], *mw[3];
  int bi = 0;
  for (int s = 0; s < 4; ++s) {
    for (int j = 0; j < depths[s]; ++j, ++bi)
      for (int k = 0; k < 13; ++k) bp[bi][k] = (const float*)d_in[p++];
    if (s < 3) {
      mg[s] = (const float*)d_in[p++];
      mb[s] = (const float*)d_in[p++];
      mw[s] = (const float*)d_in[p++];
    }
  }
  const float* norm_g = (const float*)d_in[p++];
  const float* norm_b = (const float*)d_in[p++];
  const float* head_w = (const float*)d_in[p++];
  const float* head_b = (const float*)d_in[p++];

  // --- workspace layout ----------------------------------------------------
  const long long SZ  = 32LL * 3136 * 96;   // largest [B,L,C] element count
  char* ws = (char*)d_ws;
  float*  B0   = (float*)ws;                 ws += SZ * 4;          // x (fp32)
  float*  B1   = (float*)ws;                 ws += SZ * 4;          // ln1/proj/merge scratch (fp32)
  float*  B2   = (float*)ws;                 ws += SZ * 4;          // new-x scratch (fp32)
  float*  QKV  = (float*)ws;                 ws += SZ * 3 * 4;      // qkv (fp32, rows*3C)
  __bf16* BFA  = (__bf16*)ws;                ws += SZ * 2;          // GEMM A staging (bf16)
  __bf16* HID  = (__bf16*)ws;                ws += SZ * 4 * 2;      // MLP hidden (bf16, rows*4C)
  __bf16* WREP = (__bf16*)ws;                                      // repacked weights arena

  // --- repack all GEMM weights into WMMA fragment layout -------------------
  __bf16* qkv_r[12]; __bf16* proj_r[12]; __bf16* fc1_r[12]; __bf16* fc2_r[12];
  __bf16* mw_r[3];   __bf16* head_r;
  long long woff = 0;
  auto repack = [&](const float* W, int K, int N) -> __bf16* {
    int ntiles = cdiv(N, 16), ksteps = K / 32;
    long long frags = (long long)ntiles * ksteps * 32;   // threads; 16 bf16 each
    __bf16* dst = WREP + woff;
    repack_w_kernel<<<cdiv(frags, 256), 256, 0, stream>>>(W, dst, N, ksteps, frags);
    woff += frags * 16;
    return dst;
  };
  bi = 0;
  for (int s = 0; s < 4; ++s) {
    int C = dims[s];
    for (int j = 0; j < depths[s]; ++j, ++bi) {
      qkv_r[bi]  = repack(bp[bi][2],  C,     3 * C);
      proj_r[bi] = repack(bp[bi][5],  C,     C);
      fc1_r[bi]  = repack(bp[bi][9],  C,     4 * C);
      fc2_r[bi]  = repack(bp[bi][11], 4 * C, C);
    }
    if (s < 3) mw_r[s] = repack(mw[s], 4 * C, 2 * C);
  }
  head_r = repack(head_w, 768, 1000);

  // --- patch embed + first LN ----------------------------------------------
  {
    long long total = 32LL * 56 * 56 * 96;
    patch_embed_kernel<<<cdiv(total, 256), 256, 0, stream>>>(x_in, patch_w, patch_b, B1);
    ln_kernel<<<32 * 3136, WAVE, 0, stream>>>(B1, B0, pn_g, pn_b, 96, 0);
  }

  // --- stages ---------------------------------------------------------------
  bi = 0;
  for (int s = 0; s < 4; ++s) {
    const int H = res[s], C = dims[s], nh = heads[s];
    const int L = H * H, rows = 32 * L, nw = H / 7, Bn = 32 * nw * nw;
    const long long tot = (long long)rows * C;

    for (int j = 0; j < depths[s]; ++j, ++bi) {
      const int shift = (j & 1) ? shifts[s] : 0;
      const float** P = bp[bi];

      // LN1: B0 -> B1 (fp32)
      ln_kernel<<<rows, WAVE, 0, stream>>>(B0, B1, P[0], P[1], C, 0);
      // shift + window partition: B1 -> BFA (bf16)  [Bn*49, C]
      win_gather_kernel<<<cdiv(tot, 256), 256, 0, stream>>>(B1, BFA, H, C, shift, tot);
      // qkv GEMM: BFA @ [C,3C] + b -> QKV (fp32)
      gemm_wmma_kernel<<<dim3(3 * C / 16, rows / 16), WAVE, 0, stream>>>(
          BFA, qkv_r[bi], P[3], nullptr, QKV, rows, 3 * C, C, C / 32, 0);
      // window attention: QKV -> BFA (bf16)  [Bn*49, C]
      attn_kernel<<<Bn * nh, 64, 0, stream>>>(QKV, P[4], BFA, nh, C, nw, shift, H);
      // proj GEMM: BFA @ [C,C] + b -> B1 (fp32)
      gemm_wmma_kernel<<<dim3(C / 16, rows / 16), WAVE, 0, stream>>>(
          BFA, proj_r[bi], P[6], nullptr, B1, rows, C, C, C / 32, 0);
      // window reverse + un-shift + residual: B0 + rev(B1) -> B2 (new x)
      win_scatter_add_kernel<<<cdiv(tot, 256), 256, 0, stream>>>(B0, B1, B2, H, C, shift, tot);
      // LN2: B2 -> BFA (bf16)
      ln_kernel<<<rows, WAVE, 0, stream>>>(B2, BFA, P[7], P[8], C, 1);
      // fc1 + GELU: BFA @ [C,4C] + b -> HID (bf16)
      gemm_wmma_kernel<<<dim3(4 * C / 16, rows / 16), WAVE, 0, stream>>>(
          BFA, fc1_r[bi], P[10], nullptr, HID, rows, 4 * C, C, C / 32, 1 | 2);
      // fc2 + residual: HID @ [4C,C] + b + B2 -> B0 (fp32)
      gemm_wmma_kernel<<<dim3(C / 16, rows / 16), WAVE, 0, stream>>>(
          HID, fc2_r[bi], P[12], B2, B0, rows, C, 4 * C, (4 * C) / 32, 0);
    }

    if (s < 3) {
      const int rows2 = rows / 4, C4 = 4 * C;
      const long long tot2 = (long long)rows2 * C4;
      merge_gather_kernel<<<cdiv(tot2, 256), 256, 0, stream>>>(B0, B1, H, C, tot2);
      ln_kernel<<<rows2, WAVE, 0, stream>>>(B1, BFA, mg[s], mb[s], C4, 1);
      gemm_wmma_kernel<<<dim3(2 * C / 16, rows2 / 16), WAVE, 0, stream>>>(
          BFA, mw_r[s], nullptr, nullptr, B0, rows2, 2 * C, C4, C4 / 32, 0);
    }
  }

  // --- final LN, mean pool, classifier head --------------------------------
  ln_kernel<<<32 * 49, WAVE, 0, stream>>>(B0, B1, norm_g, norm_b, 768, 0);
  mean_pool_kernel<<<cdiv(32 * 768, 256), 256, 0, stream>>>(B1, BFA, 768, 49);
  gemm_wmma_kernel<<<dim3(cdiv(1000, 16), 32 / 16), WAVE, 0, stream>>>(
      BFA, head_r, head_b, nullptr, (float*)d_out, 32, 1000, 768, 768 / 32, 0);
}
